// Biaffine_66571993088049
// MI455X (gfx1250) — matrix operand order repaired and checked by reading, not compiled, not run on hardware
//
#include <hip/hip_runtime.h>
#include <hip/hip_bf16.h>

typedef __bf16 bf16_t;
typedef __attribute__((ext_vector_type(8)))  bf16_t v8bf;
typedef __attribute__((ext_vector_type(16))) bf16_t v16bf;
typedef __attribute__((ext_vector_type(8)))  float  v8f;

#define DDIM  1024
#define K2    1152            // padded inner dim (>= 1027, multiple of 128)
#define BATCH 64
#define SEQ   512
#define LDS_STRIDE 40         // 32 + 8 pad: 80B rows -> conflict-free, 16B aligned

// ---- f32 -> bf16 helpers ---------------------------------------------------
__device__ __forceinline__ bf16_t f2bf(float f) {
  unsigned u = __builtin_bit_cast(unsigned, f);
  u += 0x7FFFu + ((u >> 16) & 1u);
  unsigned short h = (unsigned short)(u >> 16);
  return __builtin_bit_cast(bf16_t, h);
}

// pack two f32 -> {bf16(a), bf16(b)} in one dword
__device__ __forceinline__ unsigned pkbf(float a, float b) {
#if __has_builtin(__builtin_amdgcn_cvt_pk_bf16_f32)
  typedef __attribute__((ext_vector_type(2))) bf16_t v2bf;
  v2bf r = __builtin_amdgcn_cvt_pk_bf16_f32(a, b);
  return __builtin_bit_cast(unsigned, r);
#else
  unsigned ua = __builtin_bit_cast(unsigned, a);
  unsigned ub = __builtin_bit_cast(unsigned, b);
  ua += 0x7FFFu + ((ua >> 16) & 1u);    // RNE bias
  ub += 0x7FFFu + ((ub >> 16) & 1u);
  return __builtin_amdgcn_perm(ub, ua, 0x07060302);  // {ub[31:16], ua[31:16]}
#endif
}

__device__ __forceinline__ v16bf cat8(v8bf lo, v8bf hi) {
  union { v16bf v; v8bf h[2]; } u;
  u.h[0] = lo; u.h[1] = hi;
  return u.v;
}

// ---- CDNA5 async global->LDS copy (ASYNCcnt-tracked) -----------------------
__device__ __forceinline__ void async_b128(unsigned lds_off, const void* gptr) {
  unsigned long long ga = (unsigned long long)(uintptr_t)gptr;
  asm volatile("global_load_async_to_lds_b128 %0, %1, off"
               :: "v"(lds_off), "v"(ga) : "memory");
}
__device__ __forceinline__ void wait_async0() {
  asm volatile("s_wait_asynccnt 0" ::: "memory");
}

// ---------------------------------------------------------------------------
// Kernel 0: Ut[j, i] = bf16(U[i, j]) via 32x32 LDS tile transpose (coalesced)
// Ut has K2 rows (rows > 1024 are zero padding).
// ---------------------------------------------------------------------------
__global__ __launch_bounds__(256) void prep_ut(const float* __restrict__ U,
                                               bf16_t* __restrict__ Ut) {
  __shared__ float t[32][33];
  const int i0 = blockIdx.x * 32;           // inner-dim tile (0..1023)
  const int j0 = blockIdx.y * 32;           // Ut row tile   (0..1151)
  const int lx = threadIdx.x & 31;
  const int ly = threadIdx.x >> 5;          // 0..7
#pragma unroll
  for (int r = 0; r < 4; ++r) {
    int i = i0 + ly + r * 8;                // row of U (coalesced along lx=j)
    int j = j0 + lx;
    float v = (j <= DDIM) ? U[(size_t)i * (DDIM + 1) + j] : 0.0f;
    t[lx][ly + r * 8] = v;                  // transposed store
  }
  __syncthreads();
#pragma unroll
  for (int r = 0; r < 4; ++r) {
    int j = j0 + ly + r * 8;                // row of Ut
    Ut[(size_t)j * DDIM + i0 + lx] = f2bf(t[ly + r * 8][lx]);
  }
}

// ---------------------------------------------------------------------------
// Kernel 2: per flattened row x: rowterm -> A_aug cols 1025/1026,
//           pack B_aug row (H in bf16, ones, colterm, zero tail)
// ---------------------------------------------------------------------------
__global__ void prep_rows(const float* __restrict__ D, const float* __restrict__ H,
                          const float* __restrict__ W,
                          bf16_t* __restrict__ Aaug, bf16_t* __restrict__ Baug) {
  int row = blockIdx.x;
  int tid = threadIdx.x;
  const float* h = H + (size_t)row * DDIM;
  const float* d = D + (size_t)row * DDIM;
  bf16_t* brow = Baug + (size_t)row * K2;

  float hs = 0.0f, ds = 0.0f;
  for (int i = tid; i < DDIM; i += 256) {
    float hv = h[i];
    hs += hv * W[i];                 // W_h[0..1023]
    ds += d[i] * W[DDIM + 1 + i];    // W_d
    brow[i] = f2bf(hv);
  }
  __shared__ float sh[256], sd[256];
  sh[tid] = hs; sd[tid] = ds;
  __syncthreads();
  for (int s = 128; s > 0; s >>= 1) {
    if (tid < s) { sh[tid] += sh[tid + s]; sd[tid] += sd[tid + s]; }
    __syncthreads();
  }
  if (tid == 0) {
    bf16_t* arow = Aaug + (size_t)row * K2;
    arow[DDIM + 1] = f2bf(sh[0] + W[DDIM]);  // rowterm = Ha[x]·W_h  (col 1025)
    arow[DDIM + 2] = f2bf(1.0f);             // col 1026 pairs with colterm
    brow[DDIM]     = f2bf(1.0f);             // Ha ones column
    brow[DDIM + 1] = f2bf(1.0f);             // pairs with rowterm
    brow[DDIM + 2] = f2bf(sd[0]);            // colterm = D[y]·W_d
  }
  for (int i = DDIM + 3 + tid; i < K2; i += 256) brow[i] = f2bf(0.0f);
}

// ---------------------------------------------------------------------------
// WMMA GEMM: C[m,n] = sum_k A[m,k] * Bmat[n,k]
// 128x128 block tile, 8 wave32s, wave = 64x32 (4x2 frags), K step 32,
// double-buffered LDS, async global->LDS for bf16 tiles.
// ---------------------------------------------------------------------------
__global__ __launch_bounds__(256) void gemm_wmma(
    const void* __restrict__ Aptr, const bf16_t* __restrict__ Bptr,
    void* __restrict__ Cptr,
    int lda, int ldb, int ldc, int K,
    long long sA, long long sB, long long sC,
    int a_is_f32, int c_is_bf16) {
  __shared__ __align__(16) bf16_t As[2][128 * LDS_STRIDE];
  __shared__ __align__(16) bf16_t Bs[2][128 * LDS_STRIDE];

  const int tid  = threadIdx.x;
  const int lane = tid & 31;
  const int wid  = tid >> 5;
  const int lr = lane & 15;         // row/col within fragment
  const int lh = lane >> 4;         // K half selector
  const int wm = wid & 1;           // wave row (2)  -> 64 rows each
  const int wn = wid >> 1;          // wave col (4)  -> 32 cols each

  const long long m0 = (long long)blockIdx.x * 128;
  const long long n0 = (long long)blockIdx.y * 128;
  const int batch = blockIdx.z;

  const float*  Af = (const float*)Aptr  + (long long)batch * sA;
  const bf16_t* Ab = (const bf16_t*)Aptr + (long long)batch * sA;
  const bf16_t* Bb = Bptr + (long long)batch * sB;

  const int tr = tid >> 1;          // staging row 0..127
  const int th = (tid & 1) * 16;    // staging column half: 0 or 16
  const int nk = K >> 5;

  v8f acc[4][2] = {};
  float4 fr[4];                     // f32 A staging registers

  auto stageB = [&](int kt, int b) {
    const bf16_t* src = Bb + (n0 + tr) * ldb + (kt << 5) + th;
    unsigned dst = (unsigned)(uintptr_t)&Bs[b][tr * LDS_STRIDE + th];
    async_b128(dst, src);
    async_b128(dst + 16, src + 8);
  };
  auto stageAbf = [&](int kt, int b) {
    const bf16_t* src = Ab + (m0 + tr) * lda + (kt << 5) + th;
    unsigned dst = (unsigned)(uintptr_t)&As[b][tr * LDS_STRIDE + th];
    async_b128(dst, src);
    async_b128(dst + 16, src + 8);
  };
  auto loadAf32 = [&](int kt) {
    const float4* s = (const float4*)(Af + (m0 + tr) * lda + (kt << 5) + th);
    fr[0] = s[0]; fr[1] = s[1]; fr[2] = s[2]; fr[3] = s[3];
  };
  auto storeAf32 = [&](int b) {
    uint4 lo, hi;
    lo.x = pkbf(fr[0].x, fr[0].y); lo.y = pkbf(fr[0].z, fr[0].w);
    lo.z = pkbf(fr[1].x, fr[1].y); lo.w = pkbf(fr[1].z, fr[1].w);
    hi.x = pkbf(fr[2].x, fr[2].y); hi.y = pkbf(fr[2].z, fr[2].w);
    hi.z = pkbf(fr[3].x, fr[3].y); hi.w = pkbf(fr[3].z, fr[3].w);
    bf16_t* dst = &As[b][tr * LDS_STRIDE + th];
    *(uint4*)(dst)     = lo;
    *(uint4*)(dst + 8) = hi;
  };

  // prologue: stage tile 0 into buffer 0
  if (a_is_f32) { loadAf32(0); storeAf32(0); }
  else          { stageAbf(0, 0); }
  stageB(0, 0);

  for (int kt = 0; kt < nk; ++kt) {
    const int cur = kt & 1, nxt = cur ^ 1;
    wait_async0();                  // our async writes for buf `cur` landed
    __syncthreads();                // everyone's writes visible

    const bool more = (kt + 1) < nk;
    if (more) {                     // issue next-tile fetches before compute
      if (a_is_f32) loadAf32(kt + 1);
      else          stageAbf(kt + 1, nxt);
      stageB(kt + 1, nxt);
      if (kt + 2 < nk)
        __builtin_prefetch(Bb + (n0 + tr) * ldb + ((kt + 2) << 5) + th, 0, 1);
    }

    // ---- fragments + WMMA on buffer `cur` ----
    v16bf bfrag[2];
#pragma unroll
    for (int fn = 0; fn < 2; ++fn) {
      const bf16_t* p = &Bs[cur][(wn * 32 + fn * 16 + lr) * LDS_STRIDE + lh * 8];
      bfrag[fn] = cat8(*(const v8bf*)p, *(const v8bf*)(p + 16));
    }
#pragma unroll
    for (int fm = 0; fm < 4; ++fm) {
      const bf16_t* p = &As[cur][(wm * 64 + fm * 16 + lr) * LDS_STRIDE + lh * 8];
      v16bf afrag = cat8(*(const v8bf*)p, *(const v8bf*)(p + 16));
#pragma unroll
      for (int fn = 0; fn < 2; ++fn) {
        acc[fm][fn] = __builtin_amdgcn_wmma_f32_16x16x32_bf16(
            false, afrag, false, bfrag[fn], (short)0, acc[fm][fn], false, false);
      }
    }

    if (more && a_is_f32) storeAf32(nxt);   // convert+store after WMMAs issued
  }

  // ---- epilogue: C/D layout: VGPR r -> M = lh*8 + r, N = lr ----
  const int mBase = (int)m0 + wm * 64;
  const int nBase = (int)n0 + wn * 32;
  if (c_is_bf16) {
    bf16_t* Cb = (bf16_t*)Cptr + (long long)batch * sC;
#pragma unroll
    for (int fm = 0; fm < 4; ++fm)
#pragma unroll
      for (int fn = 0; fn < 2; ++fn)
#pragma unroll
        for (int r = 0; r < 8; ++r) {
          int m = mBase + fm * 16 + lh * 8 + r;
          int n = nBase + fn * 16 + lr;
          Cb[(long long)m * ldc + n] = f2bf(acc[fm][fn][r]);
        }
  } else {
    float* Cf = (float*)Cptr + (long long)batch * sC;
#pragma unroll
    for (int fm = 0; fm < 4; ++fm)
#pragma unroll
      for (int fn = 0; fn < 2; ++fn)
#pragma unroll
        for (int r = 0; r < 8; ++r) {
          int m = mBase + fm * 16 + lh * 8 + r;
          int n = nBase + fn * 16 + lr;
          Cf[(long long)m * ldc + n] = acc[fm][fn][r];
        }
  }
}

// ---------------------------------------------------------------------------
extern "C" void kernel_launch(void* const* d_in, const int* in_sizes, int n_in,
                              void* d_out, int out_size, void* d_ws, size_t ws_size,
                              hipStream_t stream) {
  (void)in_sizes; (void)n_in; (void)out_size; (void)ws_size;
  const float* D = (const float*)d_in[0];   // (64, 512, 1024)
  const float* H = (const float*)d_in[1];   // (64, 512, 1024)
  const float* U = (const float*)d_in[2];   // (1024, 1025)
  const float* W = (const float*)d_in[3];   // (2049,)
  float* out = (float*)d_out;               // (64, 512, 512)

  // workspace (bf16): Ut [K2 x 1024], Aaug [32768 x K2], Baug [32768 x K2]
  bf16_t* Ut   = (bf16_t*)d_ws;
  bf16_t* Aaug = Ut + (size_t)K2 * DDIM;
  bf16_t* Baug = Aaug + (size_t)BATCH * SEQ * K2;

  // 1) U -> Ut (transpose + bf16, zero-padded rows)
  {
    dim3 g(DDIM / 32, K2 / 32, 1);
    prep_ut<<<g, 256, 0, stream>>>(U, Ut);
  }

  // 2) GEMM1: Aaug = D @ U   (M=32768, N=K2, K=1024; f32 A, bf16 C)
  {
    dim3 g(BATCH * SEQ / 128, K2 / 128, 1);
    gemm_wmma<<<g, 256, 0, stream>>>(D, Ut, Aaug,
        DDIM, DDIM, K2, DDIM, 0LL, 0LL, 0LL, /*a_f32=*/1, /*c_bf16=*/1);
  }

  // 3) row terms (overwrites Aaug cols 1025/1026) + pack Baug
  prep_rows<<<BATCH * SEQ, 256, 0, stream>>>(D, H, W, Aaug, Baug);

  // 4) GEMM2 (batched): out[b] = Aaug[b] @ Baug[b]^T  (M=N=512, K=K2, f32 C)
  {
    dim3 g(SEQ / 128, SEQ / 128, BATCH);
    gemm_wmma<<<g, 256, 0, stream>>>(Aaug, Baug, out,
        K2, K2, SEQ, K2,
        (long long)SEQ * K2, (long long)SEQ * K2, (long long)SEQ * SEQ,
        /*a_f32=*/0, /*c_bf16=*/0);
  }
}
